// EdgeUpdatingGAT_3805341024625
// MI455X (gfx1250) — matrix operand order repaired and checked by reading, not compiled, not run on hardware
//
#include <hip/hip_runtime.h>
#include <math.h>

// ---------------- types for WMMA ----------------
typedef __attribute__((ext_vector_type(16))) _Float16 v16h;
typedef __attribute__((ext_vector_type(8)))  float    v8f;

#define NODES 50000
#define EDGES 800000
#define HOUT  96

static inline unsigned cdiv_u(long long a, long long b) { return (unsigned)((a + b - 1) / b); }

// ---------------- small utility kernels ----------------
__global__ void fill_f32(float* __restrict__ p, float v, long long n) {
  long long i = (long long)blockIdx.x * blockDim.x + threadIdx.x;
  if (i < n) p[i] = v;
}

__global__ void fill_bias(float* __restrict__ out, const float* __restrict__ b, int N) {
  long long i = (long long)blockIdx.x * blockDim.x + threadIdx.x;
  if (i < (long long)N * HOUT) out[i] = b[i % HOUT];
}

__global__ void cnt_kernel(const int* __restrict__ dst, float* __restrict__ cnt, int E) {
  int e = blockIdx.x * blockDim.x + threadIdx.x;
  if (e < E) atomicAdd(&cnt[dst[e]], 1.0f);
}

__global__ void loopea_sum(const int* __restrict__ dst, const float* __restrict__ ea,
                           float* __restrict__ loopea, int E, int Kea) {
  long long i = (long long)blockIdx.x * blockDim.x + threadIdx.x;
  if (i >= (long long)E * Kea) return;
  int e = (int)(i / Kea);
  int k = (int)(i - (long long)e * Kea);
  atomicAdd(&loopea[(long long)dst[e] * Kea + k], ea[i]);
}

__global__ void loopea_div(float* __restrict__ loopea, const float* __restrict__ cnt, int N, int Kea) {
  long long i = (long long)blockIdx.x * blockDim.x + threadIdx.x;
  if (i >= (long long)N * Kea) return;
  int n = (int)(i / Kea);
  loopea[i] /= fmaxf(cnt[n], 1.0f);
}

// wae[k] = sum_j We[k,j] * ae[j]   (collapses ep@a_e to eaa@wae)
__global__ void compute_wae(const float* __restrict__ We, const float* __restrict__ ae,
                            float* __restrict__ wae, int Kea) {
  int k = blockIdx.x * blockDim.x + threadIdx.x;
  if (k >= Kea) return;
  float s = 0.f;
  for (int j = 0; j < HOUT; ++j) s += We[(long long)k * HOUT + j] * ae[j];
  wae[k] = s;
}

// per-node dual dot products: ss = h . a_s, sd = h . a_d  (8 lanes per row)
__global__ void node_scores(const float* __restrict__ h, const float* __restrict__ as_,
                            const float* __restrict__ ad_, float* __restrict__ ss,
                            float* __restrict__ sd, int N) {
  int t = blockIdx.x * blockDim.x + threadIdx.x;
  int n = t >> 3, j = t & 7;
  if (n >= N) return;
  float a = 0.f, d = 0.f;
  for (int k = j; k < HOUT; k += 8) {
    float v = h[(long long)n * HOUT + k];
    a += v * as_[k]; d += v * ad_[k];
  }
  for (int off = 4; off; off >>= 1) { a += __shfl_xor(a, off, 8); d += __shfl_xor(d, off, 8); }
  if (j == 0) { ss[n] = a; sd[n] = d; }
}

// eae[e] = dot(eaa[e], wae) over augmented edges (8 lanes per edge)
__global__ void edge_ae(const float* __restrict__ ea, const float* __restrict__ loopea,
                        const float* __restrict__ wae, float* __restrict__ eae,
                        int E, int N, int Kea) {
  int t = blockIdx.x * blockDim.x + threadIdx.x;
  int e = t >> 3, j = t & 7;
  if (e >= E + N) return;
  const float* row = (e < E) ? (ea + (long long)e * Kea) : (loopea + (long long)(e - E) * Kea);
  float a = 0.f;
  for (int k = j; k < Kea; k += 8) a += row[k] * wae[k];
  for (int off = 4; off; off >>= 1) a += __shfl_xor(a, off, 8);
  if (j == 0) eae[e] = a;
}

__device__ inline void atomicMaxFloat(float* addr, float val) {
  if (val >= 0.f) atomicMax((int*)addr, __float_as_int(val));
  else            atomicMin((unsigned int*)addr, __float_as_uint(val));
}

__global__ void logits_max(const int* __restrict__ src, const int* __restrict__ dst,
                           const float* __restrict__ ss, const float* __restrict__ sd,
                           const float* __restrict__ eae, float* __restrict__ logit,
                           float* __restrict__ m, int E, int N) {
  int e = blockIdx.x * blockDim.x + threadIdx.x;
  if (e >= E + N) return;
  int sa = (e < E) ? src[e] : (e - E);
  int da = (e < E) ? dst[e] : (e - E);
  float l = ss[sa] + sd[da] + eae[e];
  l = (l > 0.f) ? l : 0.2f * l;
  logit[e] = l;
  atomicMaxFloat(&m[da], l);
}

__global__ void softmax_denom(const int* __restrict__ dst, const float* __restrict__ logit,
                              const float* __restrict__ m, float* __restrict__ denom, int E, int N) {
  int e = blockIdx.x * blockDim.x + threadIdx.x;
  if (e >= E + N) return;
  int da = (e < E) ? dst[e] : (e - E);
  atomicAdd(&denom[da], __expf(logit[e] - m[da]));
}

// out[da,:] += alpha * h[sa,:]   (one wave32 per augmented edge, 96 cols)
__global__ void aggregate(const int* __restrict__ src, const int* __restrict__ dst,
                          const float* __restrict__ logit, const float* __restrict__ m,
                          const float* __restrict__ denom, const float* __restrict__ h,
                          float* __restrict__ out, int E, int N) {
  int t = blockIdx.x * blockDim.x + threadIdx.x;
  int e = t >> 5, lane = t & 31;
  if (e >= E + N) return;
  int sa = (e < E) ? src[e] : (e - E);
  int da = (e < E) ? dst[e] : (e - E);
  float alpha = __expf(logit[e] - m[da]) / denom[da];
  #pragma unroll
  for (int c = lane; c < HOUT; c += 32)
    atomicAdd(&out[(long long)da * HOUT + c], alpha * h[(long long)sa * HOUT + c]);
}

// ---------------- WMMA GEMM: C[M,96] = rows(A)[M,K] @ B[K,96] (+bias) ----------------
// 8 waves/block, each wave owns a 16x96 C slab (6 f32 16x16 accumulators).
// B panel is staged to LDS PRE-SWIZZLED into WMMA fragment layout so each lane reads
// its whole v16h operand with one 32-byte LDS access (2x ds_load_b128).
//
// Fragment layout in LDS, per (ktile = k0/32, ntile = n/16):
//   halfword index = fragid*512 + lane*16 + (k%16)
//   where lane = (n%16) + 16*((k%32)/16)        [B: lanes 0-15 K=k0..+15, 16-31 K=+16..+31]

struct NodeRows {
  const float* A; int K;
  __device__ void prep(int) {}
  __device__ const float* tile_ptr(int row, int k0) const { return A + (long long)row * K + k0; }
};

struct EdgeRows {   // row e = [ x[src[e]] (96) | x[dst[e]] (96) | ea[e] (Kea) ]
  const float* xs; const int* src; const int* dst; const float* ea; int Kea;
  int se, de;
  __device__ void prep(int row) { se = src[row]; de = dst[row]; }
  // each 32-deep K tile lies entirely in one segment (96, 96, Kea all multiples of 32)
  __device__ const float* tile_ptr(int row, int k0) const {
    if (k0 < 96)  return xs + (long long)se * 96 + k0;
    if (k0 < 192) return xs + (long long)de * 96 + (k0 - 96);
    return ea + (long long)row * Kea + (k0 - 192);
  }
};

__device__ inline void cvt8(v16h& a, int base, float4 lo, float4 hi) {
  a[base + 0] = (_Float16)lo.x; a[base + 1] = (_Float16)lo.y;
  a[base + 2] = (_Float16)lo.z; a[base + 3] = (_Float16)lo.w;
  a[base + 4] = (_Float16)hi.x; a[base + 5] = (_Float16)hi.y;
  a[base + 6] = (_Float16)hi.z; a[base + 7] = (_Float16)hi.w;
}

template <typename Rows>
__global__ __launch_bounds__(256)
void wmma_gemm_kernel(Rows rows, const float* __restrict__ Bg, const float* __restrict__ bias,
                      float* __restrict__ C, int M, int K) {
  extern __shared__ _Float16 Bs[];           // (K/32)*6 fragments * 512 halves
  const int tid = threadIdx.x;
  // stage + swizzle B (f32 row-major [K,96] -> f16 fragment layout)
  for (int i = tid; i < K * HOUT; i += 256) {
    int k = i / HOUT, n = i - k * HOUT;
    int ktile = k >> 5, kin = k & 31;
    int ntile = n >> 4;
    int lane  = (n & 15) + ((kin >> 4) << 4);
    int pos   = ((ktile * 6 + ntile) << 9) + (lane << 4) + (kin & 15);
    Bs[pos] = (_Float16)Bg[i];
  }
  __syncthreads();

  const int wave = tid >> 5;
  const int lane = tid & 31;
  const int row0 = blockIdx.x * 128 + wave * 16;
  if (row0 >= M) return;

  const int lrow  = lane & 15;
  const int lhi   = lane >> 4;
  const int khalf = lhi * 16;                // A-fragment K offset (contiguous halves)
  int arow = row0 + lrow;
  if (arow >= M) arow = M - 1;               // clamp for tail; store is guarded
  rows.prep(arow);

  v8f acc[6] = {};

  for (int k0 = 0; k0 < K; k0 += 32) {
    const float* p = rows.tile_ptr(arow, k0);
    // A fragment: lanes 0-15 hold K = k0+{0..15}; lanes 16-31 K = k0+{16..31}
    float4 f0 = *reinterpret_cast<const float4*>(p + khalf);
    float4 f1 = *reinterpret_cast<const float4*>(p + khalf + 4);
    float4 f2 = *reinterpret_cast<const float4*>(p + khalf + 8);
    float4 f3 = *reinterpret_cast<const float4*>(p + khalf + 12);
    v16h a;
    cvt8(a, 0, f0, f1);
    cvt8(a, 8, f2, f3);

    const _Float16* fb = Bs + (((k0 >> 5) * 6) << 9) + (lane << 4);
    #pragma unroll
    for (int t = 0; t < 6; ++t) {
      v16h b = *reinterpret_cast<const v16h*>(fb + (t << 9));   // 32B aligned: 2x ds_load_b128
      acc[t] = __builtin_amdgcn_wmma_f32_16x16x32_f16(false, a, false, b,
                                                      (short)0, acc[t], false, false);
    }
  }

  // C layout: VGPR j -> M = j (lanes 0-15) / j+8 (lanes 16-31); N = lane&15
  #pragma unroll
  for (int t = 0; t < 6; ++t) {
    const int n = t * 16 + lrow;
    #pragma unroll
    for (int j = 0; j < 8; ++j) {
      int mrow = row0 + j + lhi * 8;
      if (mrow < M) {
        float v = acc[t][j];
        if (bias) v += bias[n];
        C[(long long)mrow * HOUT + n] = v;
      }
    }
  }
}

// ---------------- host-side layer driver ----------------
struct Ws {
  float *h, *x1, *loopea, *cnt, *ss, *sd, *m, *denom, *wae, *logit, *eae, *ea1;
};

static void gat_layer(const float* xin, int Kin, const float* ea, int Kea,
                      const int* src, const int* dst,
                      const float* W, const float* We, const float* as_, const float* ad_,
                      const float* ae_, const float* b, float* xout, const Ws& w,
                      hipStream_t s) {
  const int N = NODES, E = EDGES;
  fill_f32<<<cdiv_u(N, 256), 256, 0, s>>>(w.cnt, 0.f, N);
  fill_f32<<<cdiv_u((long long)N * Kea, 256), 256, 0, s>>>(w.loopea, 0.f, (long long)N * Kea);
  fill_f32<<<cdiv_u(N, 256), 256, 0, s>>>(w.m, -3.402823466e38f, N);
  fill_f32<<<cdiv_u(N, 256), 256, 0, s>>>(w.denom, 0.f, N);
  fill_bias<<<cdiv_u((long long)N * HOUT, 256), 256, 0, s>>>(xout, b, N);
  cnt_kernel<<<cdiv_u(E, 256), 256, 0, s>>>(dst, w.cnt, E);
  loopea_sum<<<cdiv_u((long long)E * Kea, 256), 256, 0, s>>>(dst, ea, w.loopea, E, Kea);
  loopea_div<<<cdiv_u((long long)N * Kea, 256), 256, 0, s>>>(w.loopea, w.cnt, N, Kea);
  NodeRows nr{xin, Kin};
  wmma_gemm_kernel<NodeRows><<<cdiv_u(N, 128), 256, (size_t)Kin * HOUT * sizeof(_Float16), s>>>(
      nr, W, nullptr, w.h, N, Kin);
  node_scores<<<cdiv_u((long long)N * 8, 256), 256, 0, s>>>(w.h, as_, ad_, w.ss, w.sd, N);
  compute_wae<<<cdiv_u(Kea, 128), 128, 0, s>>>(We, ae_, w.wae, Kea);
  edge_ae<<<cdiv_u((long long)(EDGES + N) * 8, 256), 256, 0, s>>>(ea, w.loopea, w.wae, w.eae, E, N, Kea);
  logits_max<<<cdiv_u(E + N, 256), 256, 0, s>>>(src, dst, w.ss, w.sd, w.eae, w.logit, w.m, E, N);
  softmax_denom<<<cdiv_u(E + N, 256), 256, 0, s>>>(dst, w.logit, w.m, w.denom, E, N);
  aggregate<<<cdiv_u((long long)(E + N) * 32, 256), 256, 0, s>>>(src, dst, w.logit, w.m, w.denom,
                                                                 w.h, xout, E, N);
}

extern "C" void kernel_launch(void* const* d_in, const int* in_sizes, int n_in,
                              void* d_out, int out_size, void* d_ws, size_t ws_size,
                              hipStream_t stream) {
  (void)in_sizes; (void)n_in; (void)out_size; (void)ws_size;
  const int N = NODES, E = EDGES;
  const float* x   = (const float*)d_in[0];          // [N,128]
  const int*   ei  = (const int*)d_in[1];            // [2,E]
  const float* ea0 = (const float*)d_in[2];          // [E,64]
  const float* w0  = (const float*)d_in[3];          // [128,96]
  const float* we0 = (const float*)d_in[4];          // [64,96]
  const float* as0 = (const float*)d_in[5];
  const float* ad0 = (const float*)d_in[6];
  const float* ae0 = (const float*)d_in[7];
  const float* b0  = (const float*)d_in[8];
  const float* ew0 = (const float*)d_in[9];          // [256,96]
  const float* eb0 = (const float*)d_in[10];
  const float* w1  = (const float*)d_in[11];         // [96,96]
  const float* we1 = (const float*)d_in[12];
  const float* as1 = (const float*)d_in[13];
  const float* ad1 = (const float*)d_in[14];
  const float* ae1 = (const float*)d_in[15];
  const float* b1  = (const float*)d_in[16];
  const float* ew1 = (const float*)d_in[17];         // [288,96]
  const float* eb1 = (const float*)d_in[18];
  const int* src = ei;
  const int* dst = ei + E;

  float* ws = (float*)d_ws;
  size_t off = 0;
  Ws w;
  w.h      = ws + off; off += (size_t)N * HOUT;
  w.x1     = ws + off; off += (size_t)N * HOUT;
  w.loopea = ws + off; off += (size_t)N * HOUT;      // max Kea = 96
  w.cnt    = ws + off; off += N;
  w.ss     = ws + off; off += N;
  w.sd     = ws + off; off += N;
  w.m      = ws + off; off += N;
  w.denom  = ws + off; off += N;
  w.wae    = ws + off; off += 128;
  w.logit  = ws + off; off += (size_t)(E + N);
  w.eae    = ws + off; off += (size_t)(E + N);
  w.ea1    = ws + off; off += (size_t)E * HOUT;

  float* xout_final = (float*)d_out;                    // [N,96]
  float* eout_final = (float*)d_out + (size_t)N * HOUT; // [E,96]

  // ---- layer 0 ----
  gat_layer(x, 128, ea0, 64, src, dst, w0, we0, as0, ad0, ae0, b0, w.x1, w, stream);
  {                                                  // ea1 = [x1[src]|x1[dst]|ea0] @ ew0 + eb0
    EdgeRows er{w.x1, src, dst, ea0, 64, 0, 0};
    const int K = 2 * HOUT + 64;                     // 256
    wmma_gemm_kernel<EdgeRows><<<cdiv_u(E, 128), 256, (size_t)K * HOUT * sizeof(_Float16), stream>>>(
        er, ew0, eb0, w.ea1, E, K);
  }
  // ---- layer 1 ----
  gat_layer(w.x1, HOUT, w.ea1, HOUT, src, dst, w1, we1, as1, ad1, ae1, b1, xout_final, w, stream);
  {                                                  // eout = [x[src]|x[dst]|ea1] @ ew1 + eb1
    EdgeRows er{xout_final, src, dst, w.ea1, HOUT, 0, 0};
    const int K = 3 * HOUT;                          // 288
    wmma_gemm_kernel<EdgeRows><<<cdiv_u(E, 128), 256, (size_t)K * HOUT * sizeof(_Float16), stream>>>(
        er, ew1, eb1, eout_final, E, K);
  }
}
